// FineGenerator_30167850287763
// MI455X (gfx1250) — matrix operand order repaired and checked by reading, not compiled, not run on hardware
//
#include <hip/hip_runtime.h>

// ---------------------------------------------------------------------------
// Contextual attention "fine generator" for MI455X (gfx1250, wave32, WMMA).
//
// Pipeline (all buffers in d_ws, everything L2-resident on a 192MB L2):
//   1) k_build_f_patches : f -> Apat (P=1024 x K=1152) bf16 hi/lo split
//   2) k_build_w_patches : b -> Bpat (normalized rows)  bf16 hi/lo split
//   3) k_compute_mm      : mask -> mm[1024]
//   4) gemm (wmma)       : scores[q][p] = Bpat . Apat^T   (per batch)
//   5) k_fuse1/k_fuse2   : diagonal fuse in both flattenings
//   6) k_softmax         : masked softmax over q, writes attn^T bf16 hi/lo
//   7) k_build_raw_w     : b -> Wm (N=2048 x K=1024) bf16 hi/lo (row-major K)
//   8) gemm (wmma)       : contrib[sp][n] = attn^T . Wm^T  (per batch)
//   9) k_gather          : 4-tap transposed-conv gather, *0.25 -> d_out
// ---------------------------------------------------------------------------

typedef __bf16 bf16;
typedef __attribute__((ext_vector_type(16))) __bf16 v16bf;
typedef __attribute__((ext_vector_type(8)))  float  v8f;

#define PQ   1024   // patch count (32*32), also Hf*Wf
#define KK1  1152   // 128 * 3 * 3
#define NN2  2048   // 128 * 4 * 4

static __device__ __forceinline__ void split_store(float v, bf16* __restrict__ hi,
                                                   bf16* __restrict__ lo, long idx) {
  bf16 h = (bf16)v;
  hi[idx] = h;
  lo[idx] = (bf16)(v - (float)h);
}

// ---------------- 1) fp patches: fd = f[:, :, ::2, ::2], 3x3 SAME patches ----
__global__ void k_build_f_patches(const float* __restrict__ f,
                                  bf16* __restrict__ hi, bf16* __restrict__ lo) {
  long idx = (long)blockIdx.x * blockDim.x + threadIdx.x;
  const long total = 2L * PQ * KK1;
  if (idx >= total) return;
  int k = (int)(idx % KK1);
  long t = idx / KK1;
  int p = (int)(t % PQ);
  int b = (int)(t / PQ);
  int c = k / 9, r = k % 9;
  int i = r / 3, j = r % 3;
  int ph = p >> 5, pw = p & 31;
  int y = ph - 1 + i, x = pw - 1 + j;
  float v = 0.f;
  if ((unsigned)y < 32u && (unsigned)x < 32u)
    v = f[(((long)b * 128 + c) * 64 + (y << 1)) * 64 + (x << 1)];
  split_store(v, hi, lo, idx);
}

// ---------------- 2) wn patches: bd 3x3 patches, row-L2-normalized -----------
__global__ void k_build_w_patches(const float* __restrict__ bsrc,
                                  bf16* __restrict__ hi, bf16* __restrict__ lo) {
  __shared__ float vals[KK1];
  __shared__ float red[128];
  int q = blockIdx.x;
  int b = blockIdx.y;
  int tid = threadIdx.x;
  int qh = q >> 5, qw = q & 31;
  float ss = 0.f;
  for (int k = tid; k < KK1; k += 128) {
    int c = k / 9, r = k % 9;
    int i = r / 3, j = r % 3;
    int y = qh - 1 + i, x = qw - 1 + j;
    float v = 0.f;
    if ((unsigned)y < 32u && (unsigned)x < 32u)
      v = bsrc[(((long)b * 128 + c) * 64 + (y << 1)) * 64 + (x << 1)];
    vals[k] = v;
    ss += v * v;
  }
  red[tid] = ss;
  __syncthreads();
  for (int s = 64; s > 0; s >>= 1) {
    if (tid < s) red[tid] += red[tid + s];
    __syncthreads();
  }
  // sum(w*w + 1e-4) over 1152 elements = sumsq + 0.1152
  float scale = rsqrtf(red[0] + 0.1152f);
  long base = ((long)b * PQ + q) * KK1;
  for (int k = tid; k < KK1; k += 128) {
    split_store(vals[k] * scale, hi, lo, base + k);
  }
}

// ---------------- 3) mm mask (from batch 0, md = mask[0,0,::8,::8]) ----------
__global__ void k_compute_mm(const float* __restrict__ mask, float* __restrict__ mm) {
  int q = blockIdx.x * blockDim.x + threadIdx.x;
  if (q >= PQ) return;
  int qh = q >> 5, qw = q & 31;
  float s = 0.f;
  for (int i = 0; i < 3; ++i)
    for (int j = 0; j < 3; ++j) {
      int y = qh - 1 + i, x = qw - 1 + j;
      if ((unsigned)y < 32u && (unsigned)x < 32u)
        s += mask[(long)(y << 3) * 256 + (x << 3)];
    }
  mm[q] = (s == 0.f) ? 1.f : 0.f;
}

// ---------------- WMMA fragment loaders (row-major-K operands) --------------
// A fragment (16x32, M=lane&15): lanes 0-15 hold K 0..15, lanes 16-31
// hold K 16..31 -> two contiguous 16B loads.
static __device__ __forceinline__ v16bf load_a_frag(const bf16* __restrict__ row, int laneHalf) {
  v16bf r;
  uint4* pr = reinterpret_cast<uint4*>(&r);
  pr[0] = *reinterpret_cast<const uint4*>(row + laneHalf * 16);
  pr[1] = *reinterpret_cast<const uint4*>(row + laneHalf * 16 + 8);
  return r;
}
// B fragment (32x16, N=lane&15): lanes 0-15 hold K 0..15, lanes 16-31 K 16..31
// of column N -> 16 contiguous K values (column n == row n of the NT operand).
static __device__ __forceinline__ v16bf load_b_frag(const bf16* __restrict__ row, int laneHalf) {
  v16bf r;
  uint4* pr = reinterpret_cast<uint4*>(&r);
  const bf16* p = row + laneHalf * 16;
  pr[0] = *reinterpret_cast<const uint4*>(p);
  pr[1] = *reinterpret_cast<const uint4*>(p + 8);
  return r;
}

// ---------------- 4)+(8) NT GEMM, bf16 split (hi,lo), f32 accumulate --------
// C[m][n] = sum_k (Ahi+Alo)[m][k] * (Bhi+Blo)[n][k], dropping lo*lo.
// One wave per 32x32 C block; 2x2 wmma tiles; 12 v_wmma per K-step.
__global__ void k_gemm_nt_bf16split(const bf16* __restrict__ Ahi, const bf16* __restrict__ Alo,
                                    const bf16* __restrict__ Bhi, const bf16* __restrict__ Blo,
                                    float* __restrict__ C, int N, int K,
                                    long strideA, long strideB, long strideC) {
  int lane = threadIdx.x;            // 0..31, all active (EXEC must be ~0)
  int laneHalf = lane >> 4;
  int laneM = lane & 15;
  int bz = blockIdx.z;
  int m0 = blockIdx.y * 32;
  int n0 = blockIdx.x * 32;
  Ahi += (long)bz * strideA;
  Alo += (long)bz * strideA;
  Bhi += (long)bz * strideB;
  Blo += (long)bz * strideB;
  C   += (long)bz * strideC;

  v8f acc[2][2];
  #pragma unroll
  for (int mi = 0; mi < 2; ++mi)
    #pragma unroll
    for (int ni = 0; ni < 2; ++ni)
      acc[mi][ni] = (v8f){0.f, 0.f, 0.f, 0.f, 0.f, 0.f, 0.f, 0.f};

  for (int k0 = 0; k0 < K; k0 += 32) {
    v16bf ah[2], al[2], bh[2], bl[2];
    #pragma unroll
    for (int mi = 0; mi < 2; ++mi) {
      const bf16* rh = Ahi + (long)(m0 + mi * 16 + laneM) * K + k0;
      const bf16* rl = Alo + (long)(m0 + mi * 16 + laneM) * K + k0;
      ah[mi] = load_a_frag(rh, laneHalf);
      al[mi] = load_a_frag(rl, laneHalf);
    }
    #pragma unroll
    for (int ni = 0; ni < 2; ++ni) {
      const bf16* rh = Bhi + (long)(n0 + ni * 16 + laneM) * K + k0;
      const bf16* rl = Blo + (long)(n0 + ni * 16 + laneM) * K + k0;
      bh[ni] = load_b_frag(rh, laneHalf);
      bl[ni] = load_b_frag(rl, laneHalf);
    }
    // Unconditional speculative prefetch of the next K-panel (safe past the
    // end: translation failures are silently dropped). -> global_prefetch_b8
    __builtin_prefetch(Ahi + (long)(m0 + laneM) * K + k0 + 32, 0, 3);
    __builtin_prefetch(Bhi + (long)(n0 + laneM) * K + k0 + 32, 0, 3);
    #pragma unroll
    for (int mi = 0; mi < 2; ++mi)
      #pragma unroll
      for (int ni = 0; ni < 2; ++ni) {
        acc[mi][ni] = __builtin_amdgcn_wmma_f32_16x16x32_bf16(
            false, ah[mi], false, bh[ni], (short)0, acc[mi][ni], false, false);
        acc[mi][ni] = __builtin_amdgcn_wmma_f32_16x16x32_bf16(
            false, ah[mi], false, bl[ni], (short)0, acc[mi][ni], false, false);
        acc[mi][ni] = __builtin_amdgcn_wmma_f32_16x16x32_bf16(
            false, al[mi], false, bh[ni], (short)0, acc[mi][ni], false, false);
      }
  }
  // C/D layout: VGPR v -> row m0+mi*16+laneHalf*8+v, col n0+ni*16+(lane&15)
  #pragma unroll
  for (int mi = 0; mi < 2; ++mi)
    #pragma unroll
    for (int ni = 0; ni < 2; ++ni) {
      int row = m0 + mi * 16 + laneHalf * 8;
      int col = n0 + ni * 16 + laneM;
      #pragma unroll
      for (int v = 0; v < 8; ++v)
        C[(long)(row + v) * N + col] = acc[mi][ni][v];
    }
}

// ---------------- 5) diagonal fuse, pass 1 (row-major flatten) ---------------
__global__ void k_fuse1(const float* __restrict__ in, float* __restrict__ out) {
  long idx = (long)blockIdx.x * blockDim.x + threadIdx.x;
  if (idx >= 2L * PQ * PQ) return;
  int c = (int)(idx & 1023);
  int r = (int)((idx >> 10) & 1023);
  int b = (int)(idx >> 20);
  const float* base = in + ((long)b << 20);
  float v = base[((long)r << 10) + c];
  if (r > 0 && c > 0)       v += base[((long)(r - 1) << 10) + (c - 1)];
  if (r < 1023 && c < 1023) v += base[((long)(r + 1) << 10) + (c + 1)];
  out[idx] = v;
}

// ---------------- 6) diagonal fuse, pass 2 (transposed flatten) --------------
static __device__ __forceinline__ float read_t(const float* __restrict__ base, int rp, int cp) {
  int q = ((rp & 31) << 5) + (rp >> 5);
  int p = ((cp & 31) << 5) + (cp >> 5);
  return base[((long)q << 10) + p];
}
__global__ void k_fuse2(const float* __restrict__ in, float* __restrict__ out) {
  long idx = (long)blockIdx.x * blockDim.x + threadIdx.x;
  if (idx >= 2L * PQ * PQ) return;
  int cp = (int)(idx & 1023);
  int rp = (int)((idx >> 10) & 1023);
  int b = (int)(idx >> 20);
  const float* base = in + ((long)b << 20);
  float v = read_t(base, rp, cp);
  if (rp > 0 && cp > 0)       v += read_t(base, rp - 1, cp - 1);
  if (rp < 1023 && cp < 1023) v += read_t(base, rp + 1, cp + 1);
  int q = ((rp & 31) << 5) + (rp >> 5);
  int p = ((cp & 31) << 5) + (cp >> 5);
  out[((long)b << 20) + ((long)q << 10) + p] = v;
}

// ---------------- 7) masked softmax over q; emit attn^T bf16 hi/lo -----------
__global__ void k_softmax(const float* __restrict__ scores, const float* __restrict__ mm,
                          bf16* __restrict__ thi, bf16* __restrict__ tlo) {
  __shared__ float red[128];
  int p = blockIdx.x;
  int b = blockIdx.y;
  int tid = threadIdx.x;
  float z[8];
  float mx = -1e30f;
  #pragma unroll
  for (int i = 0; i < 8; ++i) {
    int q = tid + i * 128;
    float m = mm[q];
    float v = scores[((long)b << 20) + ((long)q << 10) + p] * 10.0f * m; // SCALE=10
    z[i] = v;
    mx = fmaxf(mx, v);
  }
  red[tid] = mx; __syncthreads();
  for (int s = 64; s > 0; s >>= 1) {
    if (tid < s) red[tid] = fmaxf(red[tid], red[tid + s]);
    __syncthreads();
  }
  mx = red[0];
  __syncthreads();
  float sm = 0.f;
  #pragma unroll
  for (int i = 0; i < 8; ++i) {
    float e = __expf(z[i] - mx);
    z[i] = e;
    sm += e;
  }
  red[tid] = sm; __syncthreads();
  for (int s = 64; s > 0; s >>= 1) {
    if (tid < s) red[tid] += red[tid + s];
    __syncthreads();
  }
  float inv = 1.0f / red[0];
  long base = ((long)b * PQ + p) * PQ;     // attn^T[p][q]: row-major along K=q
  #pragma unroll
  for (int i = 0; i < 8; ++i) {
    int q = tid + i * 128;
    split_store(z[i] * inv * mm[q], thi, tlo, base + q);
  }
}

// ---------------- 7b) raw 4x4 patches of b, layout Wm[b][n][l] ---------------
__global__ void k_build_raw_w(const float* __restrict__ bsrc,
                              bf16* __restrict__ hi, bf16* __restrict__ lo) {
  long idx = (long)blockIdx.x * blockDim.x + threadIdx.x;
  if (idx >= 2L * NN2 * PQ) return;
  int l  = (int)(idx & 1023);
  int n  = (int)((idx >> 10) & 2047);
  int b  = (int)(idx >> 21);
  int co = n >> 4;
  int u  = (n >> 2) & 3;
  int v  = n & 3;
  int lh = l >> 5, lw = l & 31;
  int y = (lh << 1) - 1 + u;
  int x = (lw << 1) - 1 + v;
  float val = 0.f;
  if ((unsigned)y < 64u && (unsigned)x < 64u)
    val = bsrc[(((long)b * 128 + co) * 64 + y) * 64 + x];
  split_store(val, hi, lo, idx);
}

// ---------------- 9) transposed-conv gather: <=4 taps per output, /4 ---------
__global__ void k_gather(const float* __restrict__ contrib, float* __restrict__ out) {
  long idx = (long)blockIdx.x * blockDim.x + threadIdx.x;
  if (idx >= 2L * 128 * 64 * 64) return;
  int ow = (int)(idx & 63);
  int oh = (int)((idx >> 6) & 63);
  int co = (int)((idx >> 12) & 127);
  int b  = (int)(idx >> 19);
  float acc = 0.f;
  int u0 = (oh + 1) & 1;
  int v0 = (ow + 1) & 1;
  for (int u = u0; u < 4; u += 2) {
    int ih2 = oh + 1 - u;
    if (ih2 < 0) continue;
    int ih = ih2 >> 1;
    if (ih >= 32) continue;
    for (int v = v0; v < 4; v += 2) {
      int iw2 = ow + 1 - v;
      if (iw2 < 0) continue;
      int iw = iw2 >> 1;
      if (iw >= 32) continue;
      int sp = (ih << 5) + iw;
      int n  = (co << 4) + (u << 2) + v;
      acc += contrib[(((long)b * PQ) + sp) * NN2 + n];
    }
  }
  out[idx] = acc * 0.25f;
}

// ---------------------------------------------------------------------------
extern "C" void kernel_launch(void* const* d_in, const int* in_sizes, int n_in,
                              void* d_out, int out_size, void* d_ws, size_t ws_size,
                              hipStream_t stream) {
  (void)in_sizes; (void)n_in; (void)out_size;
  const float* f    = (const float*)d_in[0];   // (2,128,64,64)
  const float* bsrc = (const float*)d_in[1];   // (2,128,64,64)
  const float* mask = (const float*)d_in[2];   // (2,1,256,256)
  float* out = (float*)d_out;                  // (2,128,64,64)
  char* w = (char*)d_ws;

  // Workspace layout (~74 MB total; assumed <= ws_size)
  constexpr size_t SZ_APAT    = 2UL * PQ * KK1 * 2;      //  4,718,592 (bf16)
  constexpr size_t SZ_SCORES  = 2UL * PQ * PQ  * 4;      //  8,388,608 (f32)
  constexpr size_t SZ_ATT     = 2UL * PQ * PQ  * 2;      //  4,194,304 (bf16)
  constexpr size_t SZ_WM      = 2UL * NN2 * PQ * 2;      //  8,388,608 (bf16)
  constexpr size_t SZ_CONTRIB = 2UL * PQ * NN2 * 4;      // 16,777,216 (f32)

  size_t off = 0;
  bf16*  Apat_hi = (bf16*)(w + off);  off += SZ_APAT;
  bf16*  Apat_lo = (bf16*)(w + off);  off += SZ_APAT;
  bf16*  Bpat_hi = (bf16*)(w + off);  off += SZ_APAT;
  bf16*  Bpat_lo = (bf16*)(w + off);  off += SZ_APAT;
  float* scores  = (float*)(w + off); off += SZ_SCORES;
  float* tmp     = (float*)(w + off); off += SZ_SCORES;
  bf16*  att_hi  = (bf16*)(w + off);  off += SZ_ATT;
  bf16*  att_lo  = (bf16*)(w + off);  off += SZ_ATT;
  bf16*  wm_hi   = (bf16*)(w + off);  off += SZ_WM;
  bf16*  wm_lo   = (bf16*)(w + off);  off += SZ_WM;
  float* contrib = (float*)(w + off); off += SZ_CONTRIB;
  float* mm      = (float*)(w + off); off += 4096;
  (void)ws_size;

  // 1) fp patch matrix (unnormalized)
  k_build_f_patches<<<(unsigned)((2L * PQ * KK1 + 255) / 256), 256, 0, stream>>>(f, Apat_hi, Apat_lo);
  // 2) wn patch matrix (normalized rows)
  k_build_w_patches<<<dim3(PQ, 2), 128, 0, stream>>>(bsrc, Bpat_hi, Bpat_lo);
  // 3) mm mask
  k_compute_mm<<<4, 256, 0, stream>>>(mask, mm);
  // 4) scores[q][p] = Bpat . Apat^T  (M=N=1024, K=1152)
  k_gemm_nt_bf16split<<<dim3(PQ / 32, PQ / 32, 2), 32, 0, stream>>>(
      Bpat_hi, Bpat_lo, Apat_hi, Apat_lo, scores, PQ, KK1,
      (long)PQ * KK1, (long)PQ * KK1, (long)PQ * PQ);
  // 5) fuse
  k_fuse1<<<(unsigned)((2L * PQ * PQ + 255) / 256), 256, 0, stream>>>(scores, tmp);
  k_fuse2<<<(unsigned)((2L * PQ * PQ + 255) / 256), 256, 0, stream>>>(tmp, scores);
  // 6) softmax over q -> attn^T (bf16 hi/lo, row-major along q)
  k_softmax<<<dim3(PQ, 2), 128, 0, stream>>>(scores, mm, att_hi, att_lo);
  // 7) raw 4x4 patch matrix of b, layout [n][l]
  k_build_raw_w<<<(unsigned)((2L * NN2 * PQ + 255) / 256), 256, 0, stream>>>(bsrc, wm_hi, wm_lo);
  // 8) contrib[sp][n] = attn^T . Wm^T  (M=1024, N=2048, K=1024)
  k_gemm_nt_bf16split<<<dim3(NN2 / 32, PQ / 32, 2), 32, 0, stream>>>(
      att_hi, att_lo, wm_hi, wm_lo, contrib, NN2, PQ,
      (long)PQ * PQ, (long)NN2 * PQ, (long)PQ * NN2);
  // 9) gather + /4
  k_gather<<<(unsigned)((2L * 128 * 64 * 64 + 255) / 256), 256, 0, stream>>>(contrib, out);
}
